// joint_12730283065657
// MI455X (gfx1250) — matrix operand order repaired and checked by reading, not compiled, not run on hardware
//
#include <hip/hip_runtime.h>
#include <hip/hip_bf16.h>
#include <math.h>

typedef __bf16 bf16_t;
typedef __attribute__((ext_vector_type(16))) __bf16 v16bf;
typedef __attribute__((ext_vector_type(8)))  __bf16 v8bf;
typedef __attribute__((ext_vector_type(8)))  float  v8f;

#define BDIM 2048
#define ADIM 512
#define NHEAD 128
#define EPS_BN 1e-5f

#define WMMA_BF16(a, b, c) \
    __builtin_amdgcn_wmma_f32_16x16x32_bf16(false, (a), false, (b), (short)0, (c), false, false)

// ---------- operand loaders ----------
// A-matrix 16x32 bf16 tile from row-major [rows x ld] buffer.
// ISA layout: lanes 0-15 row M=lane, elems 0..7 = K0..7, elems 8..15 = K16..23;
// lanes 16-31 same rows, K8..15 and K24..31.
__device__ __forceinline__ v16bf load_tileA(const bf16_t* __restrict__ base, int ld,
                                            int row0, int k0, int lane) {
    int r  = row0 + (lane & 15);
    int kk = k0 + ((lane >> 4) << 3);
    const bf16_t* p = base + (size_t)r * ld + kk;
    v8bf lo = *(const v8bf*)(p);
    v8bf hi = *(const v8bf*)(p + 16);
    v16bf out;
#pragma unroll
    for (int i = 0; i < 8; ++i) { out[i] = lo[i]; out[i + 8] = hi[i]; }
    return out;
}

// B-matrix 32x16 bf16 tile, supplied as B^T row-major [cols x ld] (lane holds one column).
// Lanes 0-15: K=0..7 and K16..23; lanes 16-31: K8..15 and K24..31 (matches A).
__device__ __forceinline__ v16bf load_tileB(const bf16_t* __restrict__ baseT, int ld,
                                            int n0, int k0, int lane) {
    int r  = n0 + (lane & 15);
    int kk = k0 + ((lane >> 4) << 3);
    const bf16_t* p = baseT + (size_t)r * ld + kk;
    v8bf lo = *(const v8bf*)(p);
    v8bf hi = *(const v8bf*)(p + 16);
    v16bf out;
#pragma unroll
    for (int i = 0; i < 8; ++i) { out[i] = lo[i]; out[i + 8] = hi[i]; }
    return out;
}

__device__ __forceinline__ float elu1(float z) {
    return z > 0.f ? z : (expf(z) - 1.f);
}

// ---------- precision conversion ----------
__global__ __launch_bounds__(256) void k_convert(const float* __restrict__ in,
                                                 bf16_t* __restrict__ out, int n) {
    int i = blockIdx.x * 256 + threadIdx.x;
    if (i < n) out[i] = (bf16_t)in[i];
}

// transpose + convert: out[c*R + r] = (bf16) in[r*C + c]; blockIdx.z selects matrix
__global__ __launch_bounds__(256) void k_transpose(const float* __restrict__ in,
                                                   bf16_t* __restrict__ out,
                                                   int R, int C) {
    __shared__ float tile[32][33];
    in  += (size_t)blockIdx.z * R * C;
    out += (size_t)blockIdx.z * R * C;
    int tx = threadIdx.x, ty = threadIdx.y;
    int c = blockIdx.x * 32 + tx;
#pragma unroll
    for (int i = ty; i < 32; i += 8) {
        int r = blockIdx.y * 32 + i;
        tile[i][tx] = in[(size_t)r * C + c];
    }
    __syncthreads();
    int rr = blockIdx.y * 32 + tx;
#pragma unroll
    for (int i = ty; i < 32; i += 8) {
        int cc = blockIdx.x * 32 + i;
        out[(size_t)cc * R + rr] = (bf16_t)tile[tx][i];
    }
}

// ---------- front layer 1: act = ELU(BN(x @ W1 + b1)) -> bf16 ----------
// 64-row x 512-col tile per workgroup; wave w covers cols [w*64, w*64+64)
__global__ __launch_bounds__(256) void k_gemm1(const bf16_t* __restrict__ xb,
                                               const bf16_t* __restrict__ w1T,
                                               const float* __restrict__ b1,
                                               const float* __restrict__ g,
                                               const float* __restrict__ be,
                                               const float* __restrict__ m,
                                               const float* __restrict__ v,
                                               bf16_t* __restrict__ act) {
    const int lane = threadIdx.x & 31;
    const int wave = threadIdx.x >> 5;
    const int r0 = blockIdx.x * 64;
    const int n0 = wave * 64;

    v8f c[4][4] = {};
    for (int kk = 0; kk < ADIM; kk += 32) {
        v16bf a[4];
#pragma unroll
        for (int rb = 0; rb < 4; ++rb)
            a[rb] = load_tileA(xb, ADIM, r0 + rb * 16, kk, lane);
#pragma unroll
        for (int cb = 0; cb < 4; ++cb) {
            v16bf bt = load_tileB(w1T, ADIM, n0 + cb * 16, kk, lane);
#pragma unroll
            for (int rb = 0; rb < 4; ++rb)
                c[rb][cb] = WMMA_BF16(a[rb], bt, c[rb][cb]);
        }
    }
#pragma unroll
    for (int cb = 0; cb < 4; ++cb) {
        int n = n0 + cb * 16 + (lane & 15);
        float bias = b1[n], mu = m[n];
        float sc   = rsqrtf(v[n] + EPS_BN) * g[n];
        float bet  = be[n];
#pragma unroll
        for (int rb = 0; rb < 4; ++rb) {
#pragma unroll
            for (int e = 0; e < 8; ++e) {
                float z = (c[rb][cb][e] + bias - mu) * sc + bet;
                int row = r0 + rb * 16 + ((lane >> 4) << 3) + e;
                act[(size_t)row * ADIM + n] = (bf16_t)elu1(z);
            }
        }
    }
}

// ---------- front layer 2 (both branches) + Fourier combine + row-normalize ----------
// 32-row tile (two accumulator sets live simultaneously)
__global__ __launch_bounds__(256) void k_gemm2(const bf16_t* __restrict__ ht,
                                               const bf16_t* __restrict__ hbw,
                                               const bf16_t* __restrict__ tw2T,
                                               const bf16_t* __restrict__ bww2T,
                                               const float* __restrict__ tb2,
                                               const float* __restrict__ bwb2,
                                               const float* __restrict__ Wv,
                                               const float* __restrict__ bnei,
                                               const float* __restrict__ Av,
                                               float* __restrict__ repr_f32,
                                               bf16_t* __restrict__ repr_bf) {
    __shared__ float ss[32];
    if (threadIdx.x < 32) ss[threadIdx.x] = 0.f;
    __syncthreads();

    const int lane = threadIdx.x & 31;
    const int wave = threadIdx.x >> 5;
    const int r0 = blockIdx.x * 32;
    const int n0 = wave * 64;

    v8f ct[2][4] = {};
    for (int kk = 0; kk < ADIM; kk += 32) {
        v16bf a0 = load_tileA(ht, ADIM, r0,      kk, lane);
        v16bf a1 = load_tileA(ht, ADIM, r0 + 16, kk, lane);
#pragma unroll
        for (int cb = 0; cb < 4; ++cb) {
            v16bf bt = load_tileB(tw2T, ADIM, n0 + cb * 16, kk, lane);
            ct[0][cb] = WMMA_BF16(a0, bt, ct[0][cb]);
            ct[1][cb] = WMMA_BF16(a1, bt, ct[1][cb]);
        }
    }
    v8f cb2[2][4] = {};
    for (int kk = 0; kk < ADIM; kk += 32) {
        v16bf a0 = load_tileA(hbw, ADIM, r0,      kk, lane);
        v16bf a1 = load_tileA(hbw, ADIM, r0 + 16, kk, lane);
#pragma unroll
        for (int cb = 0; cb < 4; ++cb) {
            v16bf bt = load_tileB(bww2T, ADIM, n0 + cb * 16, kk, lane);
            cb2[0][cb] = WMMA_BF16(a0, bt, cb2[0][cb]);
            cb2[1][cb] = WMMA_BF16(a1, bt, cb2[1][cb]);
        }
    }

    // combine in place into ct; accumulate sum of squares per row
    v8f ssacc[2] = {};
#pragma unroll
    for (int cb = 0; cb < 4; ++cb) {
        int n = n0 + cb * 16 + (lane & 15);
        float tb = tb2[n], wv = Wv[n], bn = bnei[n];
        float bb = bwb2[n], av = Av[n];
#pragma unroll
        for (int rb = 0; rb < 2; ++rb) {
#pragma unroll
            for (int e = 0; e < 8; ++e) {
                float t2  = wv * (ct[rb][cb][e] + tb) + bn;
                float bwl = cb2[rb][cb][e] + bb;
                float val = av * (sinf(t2) + cosf(t2)) + 2.f * bwl;
                ct[rb][cb][e] = val;
                ssacc[rb][e] += val * val;
            }
        }
    }
    // cross-lane reduce over the 16 columns in each half, then LDS accumulate
#pragma unroll
    for (int rb = 0; rb < 2; ++rb) {
#pragma unroll
        for (int e = 0; e < 8; ++e) {
            float s = ssacc[rb][e];
            s += __shfl_xor(s, 1, 32);
            s += __shfl_xor(s, 2, 32);
            s += __shfl_xor(s, 4, 32);
            s += __shfl_xor(s, 8, 32);
            if ((lane & 15) == 0)
                atomicAdd(&ss[rb * 16 + ((lane >> 4) << 3) + e], s);
        }
    }
    __syncthreads();

#pragma unroll
    for (int cb = 0; cb < 4; ++cb) {
        int n = n0 + cb * 16 + (lane & 15);
#pragma unroll
        for (int rb = 0; rb < 2; ++rb) {
#pragma unroll
            for (int e = 0; e < 8; ++e) {
                int lr  = rb * 16 + ((lane >> 4) << 3) + e;
                int row = r0 + lr;
                float scale = 1.f / fmaxf(sqrtf(ss[lr]), 1e-12f);
                float o = ct[rb][cb][e] * scale;
                repr_f32[(size_t)row * ADIM + n] = o;
                repr_bf[(size_t)row * ADIM + n]  = (bf16_t)o;
            }
        }
    }
}

// ---------- per-head GEMM + BN + ELU + w2-dot reduction, fully fused ----------
// 64-row x 512-col tile per workgroup; halves per-head weight re-reads vs 32-row tiles
__global__ __launch_bounds__(256) void k_head(const bf16_t* __restrict__ repr_bf,
                                              const bf16_t* __restrict__ hw1T,
                                              const float* __restrict__ hb1,
                                              const float* __restrict__ hg,
                                              const float* __restrict__ hbe,
                                              const float* __restrict__ hm,
                                              const float* __restrict__ hv,
                                              const float* __restrict__ hw2,
                                              const float* __restrict__ hb2,
                                              float* __restrict__ yT) {
    __shared__ float yacc[64];
    if (threadIdx.x < 64) yacc[threadIdx.x] = 0.f;
    __syncthreads();

    const int lane = threadIdx.x & 31;
    const int wave = threadIdx.x >> 5;
    const int r0 = blockIdx.x * 64;
    const int n0 = wave * 64;
    const int head = blockIdx.y;
    const bf16_t* wT = hw1T + (size_t)head * ADIM * ADIM;

    v8f c[4][4] = {};
    for (int kk = 0; kk < ADIM; kk += 32) {
        v16bf a[4];
#pragma unroll
        for (int rb = 0; rb < 4; ++rb)
            a[rb] = load_tileA(repr_bf, ADIM, r0 + rb * 16, kk, lane);
#pragma unroll
        for (int cb = 0; cb < 4; ++cb) {
            v16bf bt = load_tileB(wT, ADIM, n0 + cb * 16, kk, lane);
#pragma unroll
            for (int rb = 0; rb < 4; ++rb)
                c[rb][cb] = WMMA_BF16(a[rb], bt, c[rb][cb]);
        }
    }

    v8f sum[4] = {};
#pragma unroll
    for (int cb = 0; cb < 4; ++cb) {
        int n = head * ADIM + n0 + cb * 16 + (lane & 15);
        float bias = hb1[n], mu = hm[n];
        float sc   = rsqrtf(hv[n] + EPS_BN) * hg[n];
        float bet  = hbe[n];
        float w2   = hw2[n];
#pragma unroll
        for (int rb = 0; rb < 4; ++rb) {
#pragma unroll
            for (int e = 0; e < 8; ++e) {
                float z = (c[rb][cb][e] + bias - mu) * sc + bet;
                sum[rb][e] += elu1(z) * w2;
            }
        }
    }
#pragma unroll
    for (int rb = 0; rb < 4; ++rb) {
#pragma unroll
        for (int e = 0; e < 8; ++e) {
            float s = sum[rb][e];
            s += __shfl_xor(s, 1, 32);
            s += __shfl_xor(s, 2, 32);
            s += __shfl_xor(s, 4, 32);
            s += __shfl_xor(s, 8, 32);
            if ((lane & 15) == 0)
                atomicAdd(&yacc[rb * 16 + ((lane >> 4) << 3) + e], s);
        }
    }
    __syncthreads();
    if (threadIdx.x < 64) {
        int row = r0 + threadIdx.x;
        yT[(size_t)row * NHEAD + head] = yacc[threadIdx.x] + hb2[head];
    }
}

extern "C" void kernel_launch(void* const* d_in, const int* in_sizes, int n_in,
                              void* d_out, int out_size, void* d_ws, size_t ws_size,
                              hipStream_t stream) {
    (void)in_sizes; (void)n_in; (void)out_size; (void)ws_size;
    const float* x     = (const float*)d_in[0];
    const float* Av    = (const float*)d_in[2];
    const float* Wv    = (const float*)d_in[3];
    const float* bnei  = (const float*)d_in[4];
    const float* t_w1  = (const float*)d_in[5];
    const float* t_b1  = (const float*)d_in[6];
    const float* t_g   = (const float*)d_in[7];
    const float* t_be  = (const float*)d_in[8];
    const float* t_m   = (const float*)d_in[9];
    const float* t_v   = (const float*)d_in[10];
    const float* t_w2  = (const float*)d_in[11];
    const float* t_b2  = (const float*)d_in[12];
    const float* bw_w1 = (const float*)d_in[13];
    const float* bw_b1 = (const float*)d_in[14];
    const float* bw_g  = (const float*)d_in[15];
    const float* bw_be = (const float*)d_in[16];
    const float* bw_m  = (const float*)d_in[17];
    const float* bw_v  = (const float*)d_in[18];
    const float* bw_w2 = (const float*)d_in[19];
    const float* bw_b2 = (const float*)d_in[20];
    const float* h_w1  = (const float*)d_in[21];
    const float* h_b1  = (const float*)d_in[22];
    const float* h_g   = (const float*)d_in[23];
    const float* h_be  = (const float*)d_in[24];
    const float* h_m   = (const float*)d_in[25];
    const float* h_v   = (const float*)d_in[26];
    const float* h_w2  = (const float*)d_in[27];
    const float* h_b2  = (const float*)d_in[28];

    // workspace layout (bf16 buffers, all 16B-aligned)
    char* ws = (char*)d_ws;
    bf16_t* xb     = (bf16_t*)ws; ws += (size_t)BDIM * ADIM * 2;
    bf16_t* tw1T   = (bf16_t*)ws; ws += (size_t)ADIM * ADIM * 2;
    bf16_t* bww1T  = (bf16_t*)ws; ws += (size_t)ADIM * ADIM * 2;
    bf16_t* tw2T   = (bf16_t*)ws; ws += (size_t)ADIM * ADIM * 2;
    bf16_t* bww2T  = (bf16_t*)ws; ws += (size_t)ADIM * ADIM * 2;
    bf16_t* hw1T   = (bf16_t*)ws; ws += (size_t)NHEAD * ADIM * ADIM * 2;
    bf16_t* ht_act = (bf16_t*)ws; ws += (size_t)BDIM * ADIM * 2;
    bf16_t* hbw_act= (bf16_t*)ws; ws += (size_t)BDIM * ADIM * 2;
    bf16_t* reprb  = (bf16_t*)ws; ws += (size_t)BDIM * ADIM * 2;

    float* yT       = (float*)d_out;                        // [2048,128]
    float* repr_out = (float*)d_out + (size_t)BDIM * NHEAD; // [2048,512]

    // Stage 0: precision conversion / transposition
    k_convert<<<(BDIM * ADIM) / 256, 256, 0, stream>>>(x, xb, BDIM * ADIM);
    dim3 tb(32, 8);
    k_transpose<<<dim3(16, 16, 1), tb, 0, stream>>>(t_w1,  tw1T,  ADIM, ADIM);
    k_transpose<<<dim3(16, 16, 1), tb, 0, stream>>>(bw_w1, bww1T, ADIM, ADIM);
    k_transpose<<<dim3(16, 16, 1), tb, 0, stream>>>(t_w2,  tw2T,  ADIM, ADIM);
    k_transpose<<<dim3(16, 16, 1), tb, 0, stream>>>(bw_w2, bww2T, ADIM, ADIM);
    k_transpose<<<dim3(16, 16, NHEAD), tb, 0, stream>>>(h_w1, hw1T, ADIM, ADIM);

    // Stage 1: front MLP layer 1 for both branches (64-row tiles)
    k_gemm1<<<BDIM / 64, 256, 0, stream>>>(xb, tw1T,  t_b1,  t_g,  t_be,  t_m,  t_v,  ht_act);
    k_gemm1<<<BDIM / 64, 256, 0, stream>>>(xb, bww1T, bw_b1, bw_g, bw_be, bw_m, bw_v, hbw_act);

    // Stage 2: layer 2 + Fourier combine + normalize -> repr (32-row tiles)
    k_gemm2<<<BDIM / 32, 256, 0, stream>>>(ht_act, hbw_act, tw2T, bww2T, t_b2, bw_b2,
                                           Wv, bnei, Av, repr_out, reprb);

    // Stage 3: 128 fused head GEMMs -> y.T (64-row tiles)
    k_head<<<dim3(BDIM / 64, NHEAD), 256, 0, stream>>>(reprb, hw1T, h_b1, h_g, h_be,
                                                       h_m, h_v, h_w2, h_b2, yT);
}
